// CAM_SE_Module_33672543601296
// MI455X (gfx1250) — matrix operand order repaired and checked by reading, not compiled, not run on hardware
//
#include <hip/hip_runtime.h>
#include <hip/hip_bf16.h>
#include <stdint.h>

// Problem constants (from reference)
#define BDIM  16
#define CDIM  512
#define HWDIM 4096
#define CHID  64          // C/8

typedef __attribute__((ext_vector_type(16))) __bf16         v16bf;
typedef __attribute__((ext_vector_type(8)))  float          v8f;
typedef __attribute__((ext_vector_type(8)))  unsigned short ush8;

union FragAB {
  struct { ush8 lo, hi; } h;   // 32 bytes = 16 bf16 elements
  v16bf v;
};

__device__ __forceinline__ unsigned short f2bf(float f) {
  union { float f; uint32_t u; } cv; cv.f = f;
  uint32_t u = cv.u;
  uint32_t r = u + 0x7FFFu + ((u >> 16) & 1u);   // round-to-nearest-even
  return (unsigned short)(r >> 16);
}

// Async global->LDS copy of 8 bytes per lane (CDNA5 VGLOBAL async op,
// tracked by ASYNCcnt). ldsoff = LDS byte offset (low 32 bits of generic
// pointer == LDS offset per ISA 10.2), gaddr = 64-bit global address.
__device__ __forceinline__ void async_g2l_b64(unsigned ldsoff, const void* gaddr) {
  asm volatile("global_load_async_to_lds_b64 %0, %1, off"
               :: "v"(ldsoff), "v"((unsigned long long)(size_t)gaddr)
               : "memory");
}

__device__ __forceinline__ void wait_async0() {
  asm volatile("s_wait_asynccnt 0" ::: "memory");
}

// ---------------------------------------------------------------------------
// Kernel 1: x (f32) -> qb (bf16), and per-(b,c) mean for the SE branch.
// ---------------------------------------------------------------------------
__global__ __launch_bounds__(256) void k_convert_mean(
    const float* __restrict__ x, unsigned short* __restrict__ qb,
    float* __restrict__ mean) {
  const int row = blockIdx.x;                     // b*C + c
  const float* xr = x + (size_t)row * HWDIM;
  unsigned short* qr = qb + (size_t)row * HWDIM;
  float s = 0.f;
  for (int i = threadIdx.x; i < HWDIM; i += 256) {
    float v = xr[i];
    s += v;
    qr[i] = f2bf(v);
  }
  __shared__ float red[256];
  red[threadIdx.x] = s;
  __syncthreads();
  for (int st = 128; st > 0; st >>= 1) {
    if (threadIdx.x < st) red[threadIdx.x] += red[threadIdx.x + st];
    __syncthreads();
  }
  if (threadIdx.x == 0) mean[row] = red[0] * (1.0f / HWDIM);
}

// ---------------------------------------------------------------------------
// Kernel 2: SE MLP  se = sigmoid(relu(mean@W1+b1)@W2+b2).  One block / batch.
// ---------------------------------------------------------------------------
__global__ __launch_bounds__(256) void k_se(
    const float* __restrict__ mean, const float* __restrict__ W1,
    const float* __restrict__ b1, const float* __restrict__ W2,
    const float* __restrict__ b2, float* __restrict__ se) {
  const int b = blockIdx.x;
  __shared__ float m[CDIM];
  __shared__ float hid[CHID];
  for (int i = threadIdx.x; i < CDIM; i += blockDim.x) m[i] = mean[b * CDIM + i];
  __syncthreads();
  if (threadIdx.x < CHID) {
    float acc = b1[threadIdx.x];
    for (int c = 0; c < CDIM; ++c) acc += m[c] * W1[c * CHID + threadIdx.x];
    hid[threadIdx.x] = acc > 0.f ? acc : 0.f;
  }
  __syncthreads();
  for (int o = threadIdx.x; o < CDIM; o += blockDim.x) {
    float acc = b2[o];
    for (int h = 0; h < CHID; ++h) acc += hid[h] * W2[h * CDIM + o];
    se[b * CDIM + o] = 1.f / (1.f + __expf(-acc));
  }
}

// ---------------------------------------------------------------------------
// Kernel 3: energy[b] = Q Q^T  (512x512, K=4096), bf16 WMMA, f32 accum.
// Block: 64(M=c) x 128(N=d) tile, 8 waves, 4 accumulators/wave sharing one
// A fragment. Double-buffered async global->LDS staging, ONE barrier/step:
//   wait_async(own buf writes) -> barrier -> issue buf^1 -> compute buf.
// ---------------------------------------------------------------------------
#define LDT 40   // padded LDS row stride (halves) for 32-wide K tiles

__global__ __launch_bounds__(256) void k_energy(
    const unsigned short* __restrict__ qb, float* __restrict__ energy) {
  const int b  = blockIdx.z;
  const int c0 = blockIdx.y * 64;
  const int d0 = blockIdx.x * 128;
  const unsigned short* Q = qb + (size_t)b * CDIM * HWDIM;

  __shared__ unsigned short At[2][64 * LDT];    // rows c0..c0+63
  __shared__ unsigned short Bt[2][128 * LDT];   // rows d0..d0+127

  const int t      = threadIdx.x;
  const int lane   = t & 31;
  const int wave   = t >> 5;
  const int laneHi = lane >> 4;
  const int lanelo = lane & 15;
  const int tm     = wave >> 1;        // 0..3  (M tile)
  const int tn0    = (wave & 1) * 4;   // 0 or 4; N tiles tn0..tn0+3

  // Per-thread staging coords: rowX = (t+256i)>>3, col = (t&7)*4 halves.
  const int cA = (t & 7) * 4;

  v8f acc[4] = {{}, {}, {}, {}};

  // Issue async staging of K-step `kk` into buffer `buf` (6 issues/thread):
  // A tile 64x32 (2/thread), B tile 128x32 (4/thread).
  auto stage = [&](int buf, int kk) {
#pragma unroll
    for (int i = 0; i < 2; ++i) {
      int row = (t + 256 * i) >> 3;
      async_g2l_b64((unsigned)(size_t)&At[buf][row * LDT + cA],
                    Q + (size_t)(c0 + row) * HWDIM + kk + cA);
    }
#pragma unroll
    for (int i = 0; i < 4; ++i) {
      int row = (t + 256 * i) >> 3;
      async_g2l_b64((unsigned)(size_t)&Bt[buf][row * LDT + cA],
                    Q + (size_t)(d0 + row) * HWDIM + kk + cA);
    }
  };

  stage(0, 0);
  int buf = 0;
  for (int kk = 0; kk < HWDIM; kk += 32) {
    wait_async0();        // this wave's writes into `buf` are in LDS
    __syncthreads();      // all waves done writing buf AND done reading buf^1
    if (kk + 32 < HWDIM) stage(buf ^ 1, kk + 32);   // overlaps compute below

    // A fragment: row M = tm*16+lanelo, K chunks at laneHi*8 and 16+laneHi*8.
    FragAB a;
    const unsigned short* arow = &At[buf][(tm * 16 + lanelo) * LDT];
    a.h.lo = *reinterpret_cast<const ush8*>(arow + laneHi * 8);
    a.h.hi = *reinterpret_cast<const ush8*>(arow + 16 + laneHi * 8);

    // B fragments: col N = lanelo (row of Bt), K chunks at laneHi*8 and 16+laneHi*8.
#pragma unroll
    for (int j = 0; j < 4; ++j) {
      FragAB bf;
      const unsigned short* brow = &Bt[buf][((tn0 + j) * 16 + lanelo) * LDT];
      bf.h.lo = *reinterpret_cast<const ush8*>(brow + laneHi * 8);
      bf.h.hi = *reinterpret_cast<const ush8*>(brow + 16 + laneHi * 8);
      acc[j] = __builtin_amdgcn_wmma_f32_16x16x32_bf16(false, a.v, false, bf.v,
                                                       (short)0, acc[j], false, false);
    }
    buf ^= 1;
  }

  // C/D layout: lane holds (M = laneHi*8 + r, N = lanelo) for r=0..7.
  float* E = energy + ((size_t)b * CDIM + c0 + tm * 16 + laneHi * 8) * CDIM + d0;
#pragma unroll
  for (int r = 0; r < 8; ++r) {
#pragma unroll
    for (int j = 0; j < 4; ++j)
      E[(size_t)r * CDIM + (tn0 + j) * 16 + lanelo] = acc[j][r];
  }
}

// ---------------------------------------------------------------------------
// Kernel 4: attention = softmax(max - energy) == softmax(-energy)  (per row),
// output bf16. One block per (b,c) row of 512.
// ---------------------------------------------------------------------------
__global__ __launch_bounds__(256) void k_softmax(
    const float* __restrict__ energy, unsigned short* __restrict__ att) {
  const int row = blockIdx.x;                 // b*C + c
  const float* er = energy + (size_t)row * CDIM;
  __shared__ float red[256];
  const int t = threadIdx.x;
  float e0 = er[t], e1 = er[t + 256];
  red[t] = fminf(e0, e1);
  __syncthreads();
  for (int st = 128; st > 0; st >>= 1) {
    if (t < st) red[t] = fminf(red[t], red[t + st]);
    __syncthreads();
  }
  float m = red[0];
  __syncthreads();
  float p0 = __expf(m - e0), p1 = __expf(m - e1);
  red[t] = p0 + p1;
  __syncthreads();
  for (int st = 128; st > 0; st >>= 1) {
    if (t < st) red[t] += red[t + st];
    __syncthreads();
  }
  float inv = 1.0f / red[0];
  att[(size_t)row * CDIM + t]       = f2bf(p0 * inv);
  att[(size_t)row * CDIM + t + 256] = f2bf(p1 * inv);
}

// ---------------------------------------------------------------------------
// Kernel 5: out = gamma * (att @ Q) * se + x.  bf16 WMMA, fused epilogue.
// 64(M) x 128(N) tile; 8 waves; 4 N-tiles/wave sharing one A fragment.
// Double-buffered, one barrier/step: A tile staged async; B tile (needs
// transpose: contraction dim d is the slow axis of Q) staged through VGPRs.
// ---------------------------------------------------------------------------
#define LDB 40   // padded LDS row stride (halves) for transposed B tile

__global__ __launch_bounds__(256) void k_attn_out(
    const unsigned short* __restrict__ att, const unsigned short* __restrict__ qb,
    const float* __restrict__ se, const float* __restrict__ gamma,
    const float* __restrict__ x, float* __restrict__ out) {
  const int b  = blockIdx.z;
  const int c0 = blockIdx.y * 64;
  const int n0 = blockIdx.x * 128;

  const unsigned short* A = att + (size_t)b * CDIM * CDIM;
  const unsigned short* Q = qb + (size_t)b * CDIM * HWDIM;

  __shared__ unsigned short At[2][64 * LDT];
  __shared__ unsigned short BT[2][128 * LDB];   // transposed: BT[n][k]

  const int t      = threadIdx.x;
  const int lane   = t & 31;
  const int wave   = t >> 5;
  const int laneHi = lane >> 4;
  const int lanelo = lane & 15;
  const int tm     = wave >> 1;        // 0..3
  const int tn0    = (wave & 1) * 4;   // 0 or 4; tiles tn0..tn0+3

  const int cA = (t & 7) * 4;

  v8f acc[4] = {{}, {}, {}, {}};

  // Stage K-step kk into buffer buf: A tile async, B tile transposed via VGPRs.
  auto stage = [&](int buf, int kk) {
#pragma unroll
    for (int i = 0; i < 2; ++i) {
      int row = (t + 256 * i) >> 3;
      async_g2l_b64((unsigned)(size_t)&At[buf][row * LDT + cA],
                    A + (size_t)(c0 + row) * CDIM + kk + cA);
    }
#pragma unroll
    for (int i = 0; i < 8; ++i) {
      int linear = t + 256 * i;            // 32 k-rows * 64 n-dwords
      int k  = linear >> 6;
      int nd = (linear & 63) * 2;
      unsigned int pair =
          *reinterpret_cast<const unsigned int*>(Q + (size_t)(kk + k) * HWDIM + n0 + nd);
      BT[buf][nd * LDB + k]       = (unsigned short)(pair & 0xFFFFu);
      BT[buf][(nd + 1) * LDB + k] = (unsigned short)(pair >> 16);
    }
  };

  stage(0, 0);
  int buf = 0;
  for (int kk = 0; kk < CDIM; kk += 32) {
    wait_async0();
    __syncthreads();      // buf fully staged (async + ds); buf^1 free to write
    if (kk + 32 < CDIM) stage(buf ^ 1, kk + 32);

    FragAB a;
    const unsigned short* arow = &At[buf][(tm * 16 + lanelo) * LDT];
    a.h.lo = *reinterpret_cast<const ush8*>(arow + laneHi * 8);
    a.h.hi = *reinterpret_cast<const ush8*>(arow + 16 + laneHi * 8);

#pragma unroll
    for (int j = 0; j < 4; ++j) {
      FragAB bf;
      const unsigned short* brow = &BT[buf][((tn0 + j) * 16 + lanelo) * LDB];
      bf.h.lo = *reinterpret_cast<const ush8*>(brow + laneHi * 8);
      bf.h.hi = *reinterpret_cast<const ush8*>(brow + 16 + laneHi * 8);
      acc[j] = __builtin_amdgcn_wmma_f32_16x16x32_bf16(false, a.v, false, bf.v,
                                                       (short)0, acc[j], false, false);
    }
    buf ^= 1;
  }

  const float g = gamma[0];
#pragma unroll
  for (int r = 0; r < 8; ++r) {
    int c = c0 + tm * 16 + laneHi * 8 + r;
    float s = g * se[b * CDIM + c];
    const float* xr = x   + ((size_t)b * CDIM + c) * HWDIM + n0;
    float*      orw = out + ((size_t)b * CDIM + c) * HWDIM + n0;
#pragma unroll
    for (int j = 0; j < 4; ++j) {
      int n = (tn0 + j) * 16 + lanelo;
      orw[n] = s * acc[j][r] + xr[n];
    }
  }
}

// ---------------------------------------------------------------------------
// Launch
// ---------------------------------------------------------------------------
extern "C" void kernel_launch(void* const* d_in, const int* in_sizes, int n_in,
                              void* d_out, int out_size, void* d_ws, size_t ws_size,
                              hipStream_t stream) {
  const float* x     = (const float*)d_in[0];
  const float* gamma = (const float*)d_in[1];
  const float* W1    = (const float*)d_in[2];
  const float* b1    = (const float*)d_in[3];
  const float* W2    = (const float*)d_in[4];
  const float* b2    = (const float*)d_in[5];
  float* out = (float*)d_out;

  // Workspace layout (bytes)
  char* ws = (char*)d_ws;
  unsigned short* qb   = (unsigned short*)(ws);                 // 64 MiB bf16 Q
  unsigned short* attb = (unsigned short*)(ws + 67108864);      //  8 MiB bf16 attention
  float* energy        = (float*)(ws + 75497472);               // 16 MiB f32
  float* mean          = (float*)(ws + 92274688);               // 32 KiB
  float* se            = (float*)(ws + 92307456);               // 32 KiB

  k_convert_mean<<<BDIM * CDIM, 256, 0, stream>>>(x, qb, mean);
  k_se<<<BDIM, 256, 0, stream>>>(mean, W1, b1, W2, b2, se);

  dim3 ge(CDIM / 128, CDIM / 64, BDIM);       // 4 x 8 x 16 (N=d x M=c x B)
  k_energy<<<ge, 256, 0, stream>>>(qb, energy);

  k_softmax<<<BDIM * CDIM, 256, 0, stream>>>(energy, attb);

  dim3 go(HWDIM / 128, CDIM / 64, BDIM);      // 32 x 8 x 16
  k_attn_out<<<go, 256, 0, stream>>>(attb, qb, se, gamma, x, out);
}
